// KeyValueMemory_39204461478014
// MI455X (gfx1250) — matrix operand order repaired and checked by reading, not compiled, not run on hardware
//
#include <hip/hip_runtime.h>

// ---------------------------------------------------------------------------
// KeyValueMemory for MI455X (gfx1250, wave32):
//   sim = normalize(q) . normalize(k)^T -> fused top-8 -> softmax -> gather V
// * v_wmma_f32_16x16x32_f16 for the 52.6-GFLOP similarity GEMM (f16 operands
//   of unit vectors, fp32 accumulate)
// * Tensor Data Mover (tensor_load_to_lds) double-buffered key staging with
//   hardware row padding (272B LDS rows), tracked by TENSORcnt
// * top-8 / softmax / V-gather fused in-register per wave
// ---------------------------------------------------------------------------

typedef _Float16 v16h __attribute__((ext_vector_type(16)));
typedef _Float16 v8h  __attribute__((ext_vector_type(8)));
typedef float    v8f  __attribute__((ext_vector_type(8)));
typedef unsigned int u32x4 __attribute__((ext_vector_type(4)));
typedef int          i32x8 __attribute__((ext_vector_type(8)));
typedef int          i32x4 __attribute__((ext_vector_type(4)));

#define NB     32
#define NC     128
#define NHW    3136            // 56*56
#define MKEYS  2048
#define NTOPK  8
#define CHUNK_ROWS   128       // keys staged per LDS chunk (32KB payload)
#define ROW_STRIDE_B 272       // 128 halves * 2B + 16B pad (bank de-aliasing)
#define BUF_BYTES    (CHUNK_ROWS * ROW_STRIDE_B)   // 34816
#define NCHUNK       (MKEYS / CHUNK_ROWS)          // 16
#define WAVES_PER_BLOCK 4

// ---------------- sorted-descending top-8 insert (registers only) -----------
__device__ __forceinline__ void topk_insert(float (&tv)[NTOPK], int (&ti)[NTOPK],
                                            float v, int idx) {
  if (v <= tv[NTOPK - 1]) return;
  tv[NTOPK - 1] = v; ti[NTOPK - 1] = idx;
#pragma unroll
  for (int j = NTOPK - 1; j > 0; --j) {
    bool sw = tv[j] > tv[j - 1];
    float hv = sw ? tv[j] : tv[j - 1];
    float lv = sw ? tv[j - 1] : tv[j];
    int   hi = sw ? ti[j] : ti[j - 1];
    int   li = sw ? ti[j - 1] : ti[j];
    tv[j - 1] = hv; tv[j] = lv; ti[j - 1] = hi; ti[j] = li;
  }
}

// ---------------- TDM: one 128x128-f16 chunk -> padded LDS rows --------------
// D# per CDNA5 ISA: 2D tensor, data_size=2B,
// tensor_dim0=128, stride0=128, tile 128x128, pad 4 dwords every 64 dwords
// (256B row -> 272B LDS stride). Issued wave-uniformly; TENSORcnt-tracked.
__device__ __forceinline__ void tdm_load_keys_chunk(const _Float16* gsrc,
                                                    unsigned lds_off) {
  const unsigned long long ga = (unsigned long long)(size_t)gsrc;
  u32x4 g0;
  g0[0] = 1u;                                   // count=1, user descriptor
  g0[1] = lds_off;                              // lds_addr (bytes)
  g0[2] = (unsigned)(ga & 0xFFFFFFFFu);         // global_addr[31:0]
  g0[3] = (unsigned)((ga >> 32) & 0x01FFFFFFu)  // global_addr[56:32]
        | (2u << 30);                           // type = 2 ("image")

  // group1 bitfields (bits within 256-bit group, little-endian u64 lanes):
  //  [17:16] data_size=1 (2B)  [20] pad_enable  [24:22] pad_interval=5 (64dw)
  //  [31:25] pad_amount=3 (4dw)  [79:48] tensor_dim0=128
  //  [111:80] tensor_dim1=2048  [127:112] tile_dim0=128
  //  [143:128] tile_dim1=128    [207:160] tensor_dim0_stride=128
  const unsigned long long u0 = (1ull << 16) | (1ull << 20) | (5ull << 22)
                              | (3ull << 25) | (128ull << 48);
  const unsigned long long u1 = (2048ull << 16) | (128ull << 48);
  const unsigned long long u2 = 128ull | (128ull << 32);
  const unsigned long long u3 = 0ull;
  i32x8 g1;
  g1[0] = (int)u0; g1[1] = (int)(u0 >> 32);
  g1[2] = (int)u1; g1[3] = (int)(u1 >> 32);
  g1[4] = (int)u2; g1[5] = (int)(u2 >> 32);
  g1[6] = (int)u3; g1[7] = (int)(u3 >> 32);
  const i32x4 z4 = {0, 0, 0, 0};
  const i32x8 z8 = {0, 0, 0, 0, 0, 0, 0, 0};
  __builtin_amdgcn_tensor_load_to_lds(g0, g1, z4, z4, z8, 0);
}

// ---------------- kernel 1: normalize keys -> f16 row-major -----------------
__global__ __launch_bounds__(256) void normkeys_kernel(
    const float* __restrict__ K, _Float16* __restrict__ Kn) {
  const int lane = threadIdx.x & 31;
  const int m    = blockIdx.x * 8 + (threadIdx.x >> 5);
  float4 kv = ((const float4*)(K + (size_t)m * NC))[lane];
  float s = kv.x * kv.x + kv.y * kv.y + kv.z * kv.z + kv.w * kv.w;
#pragma unroll
  for (int off = 16; off >= 1; off >>= 1) s += __shfl_xor(s, off);
  const float inv = 1.f / fmaxf(sqrtf(s), 1e-12f);
  _Float16* d = Kn + (size_t)m * NC + lane * 4;
  d[0] = (_Float16)(kv.x * inv);
  d[1] = (_Float16)(kv.y * inv);
  d[2] = (_Float16)(kv.z * inv);
  d[3] = (_Float16)(kv.w * inv);
}

// ---------------- kernel 2: fused sim-GEMM + top8 + softmax + gather --------
__global__ __launch_bounds__(WAVES_PER_BLOCK * 32) void kv_topk_kernel(
    const float* __restrict__ Q,      // [32,128,56,56]
    const _Float16* __restrict__ Kn,  // [2048,128] normalized f16
    const float* __restrict__ V,      // [2048,128]
    float* __restrict__ Out) {        // [32,128,56,56]
  // two TDM key buffers; per-wave output tile overlays buffer 0 at the end
  __shared__ __align__(16) unsigned char smem[2 * BUF_BYTES];

  const int tid  = threadIdx.x;
  const int lane = tid & 31;
  const int wave = tid >> 5;
  const int half = lane >> 4;   // 0: lanes 0-15, 1: lanes 16-31
  const int qsub = lane & 15;

  const int tile = blockIdx.x * WAVES_PER_BLOCK + wave;  // 6272 tiles exactly
  const int q0   = tile * 16;
  const int b    = q0 / NHW;
  const int hw0  = q0 - b * NHW;

  const unsigned lds_base = (unsigned)(size_t)(&smem[0]);  // low 32b = LDS off

  // prologue: DMA chunk 0 into buffer 0 while we load/normalize the queries
  if (wave == 0) tdm_load_keys_chunk(Kn, lds_base);

  // ---- load 16x128 query tile, L2-normalize, pack f16 B fragments ----------
  // B-matrix layout (32x16 f16): lane = column N (query), halves h=0..15 hold
  // K = 32*t + 16*half + h  -> contiguous channels of one query.
  const float* qbase = Q + (size_t)b * NC * NHW + hw0 + qsub;
  float qv[4][16];
  float ss = 0.f;
#pragma unroll
  for (int t = 0; t < 4; ++t) {
    const int cbase = 32 * t + 16 * half;
#pragma unroll
    for (int h = 0; h < 16; ++h) {
      float x = qbase[(size_t)(cbase + h) * NHW];
      qv[t][h] = x;
      ss += x * x;
    }
  }
  ss += __shfl_xor(ss, 16);   // lane pair (L, L+16) covers all 128 channels
  const float qinv = 1.f / fmaxf(sqrtf(ss), 1e-12f);
  v16h bfrag[4];
#pragma unroll
  for (int t = 0; t < 4; ++t)
#pragma unroll
    for (int h = 0; h < 16; ++h)
      bfrag[t][h] = (_Float16)(qv[t][h] * qinv);

  // ---- per-lane running top-8 ----------------------------------------------
  float tv[NTOPK]; int ti[NTOPK];
#pragma unroll
  for (int j = 0; j < NTOPK; ++j) { tv[j] = -3.4e38f; ti[j] = 0; }

  const unsigned abase_row = (unsigned)qsub * ROW_STRIDE_B;

  for (int chunk = 0; chunk < NCHUNK; ++chunk) {
    const unsigned bufoff = (unsigned)(chunk & 1) * BUF_BYTES;
    if (wave == 0) __builtin_amdgcn_s_wait_tensorcnt(0);  // chunk's DMA done
    __syncthreads();  // buffer ready; other buffer no longer being read
    // overlap: DMA next chunk into the other buffer while we compute
    if (wave == 0 && chunk + 1 < NCHUNK)
      tdm_load_keys_chunk(Kn + (size_t)(chunk + 1) * CHUNK_ROWS * NC,
                          lds_base + (bufoff ^ BUF_BYTES));

    for (int kt = 0; kt < CHUNK_ROWS / 16; ++kt) {
      const unsigned abase = bufoff + (unsigned)(kt * 16) * ROW_STRIDE_B + abase_row;
      // A-matrix layout (16x32 f16): lane = row M (key); halves 0-15 hold
      // K = 32t + 16*half + h (contiguous, matching B).
      v16h afrag[4];
#pragma unroll
      for (int t = 0; t < 4; ++t) {
        const v8h lo = *(const v8h*)(smem + abase + (unsigned)(64 * t + 32 * half));
        const v8h hi = *(const v8h*)(smem + abase + (unsigned)(64 * t + 32 * half + 16));
#pragma unroll
        for (int h = 0; h < 8; ++h) { afrag[t][h] = lo[h]; afrag[t][h + 8] = hi[h]; }
      }
      v8f acc = {0.f, 0.f, 0.f, 0.f, 0.f, 0.f, 0.f, 0.f};
#pragma unroll
      for (int t = 0; t < 4; ++t)
        acc = __builtin_amdgcn_wmma_f32_16x16x32_f16(
            false, afrag[t], false, bfrag[t], (short)0, acc, false, false);

      // D layout: lane = query column; VGPR j = key 8*half + j of this tile
      const int keybase = chunk * CHUNK_ROWS + kt * 16 + 8 * half;
#pragma unroll
      for (int j = 0; j < 8; ++j) topk_insert(tv, ti, acc[j], keybase + j);
    }
  }

  // ---- merge partner lane's top-8 (other 8 keys of every tile) -------------
  float pv[NTOPK]; int pi[NTOPK];
#pragma unroll
  for (int j = 0; j < NTOPK; ++j) {
    pv[j] = __shfl_xor(tv[j], 16);
    pi[j] = __shfl_xor(ti[j], 16);
  }
#pragma unroll
  for (int j = 0; j < NTOPK; ++j) topk_insert(tv, ti, pv[j], pi[j]);
  // lanes 0-15 now hold the global top-8 for query q0+lane

  // ---- softmax over the 8 --------------------------------------------------
  float wgt[NTOPK];
  {
    const float mx = tv[0];  // sorted descending
    float s = 0.f;
#pragma unroll
    for (int j = 0; j < NTOPK; ++j) { wgt[j] = __expf(tv[j] - mx); s += wgt[j]; }
    const float is = 1.f / s;
#pragma unroll
    for (int j = 0; j < NTOPK; ++j) wgt[j] *= is;
  }

  __syncthreads();  // all waves done with key buffers; overlay output tiles
  float* obuf = (float*)(smem + wave * (16 * 132 * 4));

  // ---- weighted gather of V rows: broadcast each query's (w,idx) -----------
  const float4* V4 = (const float4*)V;
  for (int qq = 0; qq < 16; ++qq) {
    float4 acc = make_float4(0.f, 0.f, 0.f, 0.f);
#pragma unroll
    for (int j = 0; j < NTOPK; ++j) {
      const float wj = __shfl(wgt[j], qq);
      const int   ij = __shfl(ti[j], qq);
      const float4 vv = V4[ij * (NC / 4) + lane];  // coalesced 512B row
      acc.x = fmaf(wj, vv.x, acc.x);
      acc.y = fmaf(wj, vv.y, acc.y);
      acc.z = fmaf(wj, vv.z, acc.z);
      acc.w = fmaf(wj, vv.w, acc.w);
    }
    *(float4*)(obuf + qq * 132 + lane * 4) = acc;   // padded row: no conflicts
  }

  // ---- coalesced store: for fixed channel c, 16 queries are contiguous -----
  float* obase = Out + (size_t)b * NC * NHW + hw0;
  for (int i = lane; i < 16 * NC; i += 32) {
    const int c = i >> 4, qq = i & 15;
    obase[(size_t)c * NHW + qq] = obuf[qq * 132 + c];
  }
}

// ---------------------------------------------------------------------------
extern "C" void kernel_launch(void* const* d_in, const int* in_sizes, int n_in,
                              void* d_out, int out_size, void* d_ws, size_t ws_size,
                              hipStream_t stream) {
  (void)in_sizes; (void)n_in; (void)out_size; (void)ws_size;
  const float* q      = (const float*)d_in[0];  // [32,128,56,56]
  const float* keys   = (const float*)d_in[1];  // [2048,128]
  const float* values = (const float*)d_in[2];  // [2048,128]
  float* out = (float*)d_out;
  _Float16* kn = (_Float16*)d_ws;               // 512 KB normalized-key cache

  normkeys_kernel<<<MKEYS / 8, 256, 0, stream>>>(keys, kn);

  const int tiles  = (NB * NHW) / 16;           // 6272
  const int blocks = tiles / WAVES_PER_BLOCK;   // 1568
  kv_topk_kernel<<<blocks, WAVES_PER_BLOCK * 32, 0, stream>>>(q, kn, values, out);
}
